// IMGPCEnDecoder_23922967838815
// MI455X (gfx1250) — matrix operand order repaired and checked by reading, not compiled, not run on hardware
//
#include <hip/hip_runtime.h>
#include <hip/hip_bf16.h>
#include <math.h>

// ---------------------------------------------------------------------------
// IMGPCEnDecoder for MI455X (gfx1250).  B=16, F=128, L=S=5120.
// All GEMM-like work via v_wmma_f32_16x16x32_f16, LDS tiles stored directly
// in WMMA fragment order so fragment loads are ds_load_b128s.
// ---------------------------------------------------------------------------

typedef __attribute__((ext_vector_type(16))) _Float16 v16h;
typedef __attribute__((ext_vector_type(8)))  _Float16 v8h;
typedef __attribute__((ext_vector_type(8)))  float    v8f;

#define BDIM   16
#define FDIM   128
#define NHEADS 8
#define HDIM   16
#define IMG_H  40
#define IMG_W  128
#define LSEQ   5120
#define PPT    256
#define PIMG   320

#define BM 128            // block M tile
#define BN 64             // block N tile
#define BK 32             // K step
// LDS tiles live in fragment order:
//   lA[slab(8)][lane(32)][j(16)]  : A fragment for rows slab*16..+15
//   lB[nsub(4)][lane(32)][j(16)]  : B fragment for cols nsub*16..+15

// fragment<->tile mapping (ISA 7.12.2):
//   A: value(lane,j) = Atile[lane&15][ 16*(j>>3) + 8*(lane>>4) + (j&7) ]
//   B: value(lane,j) = Btile[ 16*(j>>3) + 8*(lane>>4) + (j&7) ][ (lane&15) ]

// ===========================================================================
// GEMM: C[b] = epi( A[b](MxK) * W(KxN) );  M=LSEQ, M%128==0, N%64==0, K%32==0
//   NK    : weights stored [out,in] (ldb=K) vs [in,out] (ldb=N)
//   HASA2 : A is concat view, k>=ksplit comes from A2 (ksplit%16==0)
//   ACT   : 0 none, 1 relu, 2 elu+1
// ===========================================================================
template<bool NK, bool AFFINE, bool HASRES, bool HASA2, int ACT>
__global__ __launch_bounds__(256) void gemm_wmma(
    const float* __restrict__ A,  long long aBatch, int lda,
    const float* __restrict__ A2, long long a2Batch, int lda2, int ksplit,
    const float* __restrict__ Wt, int ldb,
    float* __restrict__ C, long long cBatch, int ldc,
    int Ktot,
    const float* __restrict__ scale, const float* __restrict__ bias,
    const float* __restrict__ res, long long resBatch, int ldres)
{
  __shared__ alignas(32) _Float16 lA[8 * 32 * 16];   // 8 KB
  __shared__ alignas(32) _Float16 lB[4 * 32 * 16];   // 4 KB

  const int b     = blockIdx.z;
  const int tileM = blockIdx.x * BM;
  const int tileN = blockIdx.y * BN;

  const float* Ab  = A + (long long)b * aBatch;
  const float* A2b = HASA2 ? A2 + (long long)b * a2Batch : nullptr;
  float*       Cb  = C + (long long)b * cBatch;
  const float* Rb  = HASRES ? res + (long long)b * resBatch : nullptr;

  const int tid  = threadIdx.x;
  const int lane = tid & 31;
  const int wid  = tid >> 5;
  const int wm   = wid >> 1;              // 0..3 -> 32-row slab pair
  const int wn   = wid & 1;               // 0..1 -> 32-col slab pair
  const int half = lane >> 4;
  const int l16  = lane & 15;

  v8f acc00 = {}, acc01 = {}, acc10 = {}, acc11 = {};

  // A staging: thread -> row m=tid>>1 (0..127), 16 k's at ako
  const int arow = tid >> 1;
  const int ako  = (tid & 1) * 16;
  const int slab = arow >> 4, row16 = arow & 15;
  // B staging: thread -> col n=tid>>2 (0..63), 8 k's at bkc
  const int bn  = tid >> 2;
  const int bkc = (tid & 3) * 8;
  const int bnsub = bn >> 4;
  const int blane = 16 * ((bkc >> 3) & 1) + (bn & 15);
  const int bj0   = (bkc & 16) >> 1;

  for (int kk = 0; kk < Ktot; kk += BK) {
    // ---- stage A (BM x BK), f32 -> f16, fragment order ----
    {
      const int kg0 = kk + ako;
      const float* src;
      if (HASA2 && kg0 >= ksplit) src = A2b + (long long)(tileM + arow) * lda2 + (kg0 - ksplit);
      else                        src = Ab  + (long long)(tileM + arow) * lda  + kg0;
      __builtin_prefetch(src + BK, 0, 1);
      const int g = ako >> 4;
      v8h h0, h1;
#pragma unroll
      for (int i = 0; i < 8; ++i) h0[i] = (_Float16)src[i];
#pragma unroll
      for (int i = 0; i < 8; ++i) h1[i] = (_Float16)src[8 + i];
      *(v8h*)&lA[((slab * 32 + row16)      * 16) + 8 * g] = h0;  // half 0
      *(v8h*)&lA[((slab * 32 + 16 + row16) * 16) + 8 * g] = h1;  // half 1
    }
    // ---- stage B (BK x BN), fragment order ----
    {
      const int kg = kk + bkc;
      v8h hb;
#pragma unroll
      for (int i = 0; i < 8; ++i) {
        const float v = NK ? Wt[(long long)(tileN + bn) * ldb + (kg + i)]
                           : Wt[(long long)(kg + i) * ldb + (tileN + bn)];
        hb[i] = (_Float16)v;
      }
      *(v8h*)&lB[((bnsub * 32 + blane) * 16) + bj0] = hb;
    }
    __syncthreads();

    // ---- fragments: contiguous 32B per lane -> ds_load_b128 pairs ----
    const v16h a0 = *(const v16h*)&lA[(((wm * 2 + 0) * 32) + lane) * 16];
    const v16h a1 = *(const v16h*)&lA[(((wm * 2 + 1) * 32) + lane) * 16];
    const v16h b0 = *(const v16h*)&lB[(((wn * 2 + 0) * 32) + lane) * 16];
    const v16h b1 = *(const v16h*)&lB[(((wn * 2 + 1) * 32) + lane) * 16];
    acc00 = __builtin_amdgcn_wmma_f32_16x16x32_f16(false, a0, false, b0, (short)0, acc00, false, false);
    acc01 = __builtin_amdgcn_wmma_f32_16x16x32_f16(false, a0, false, b1, (short)0, acc01, false, false);
    acc10 = __builtin_amdgcn_wmma_f32_16x16x32_f16(false, a1, false, b0, (short)0, acc10, false, false);
    acc11 = __builtin_amdgcn_wmma_f32_16x16x32_f16(false, a1, false, b1, (short)0, acc11, false, false);
    __syncthreads();
  }

  // ---- epilogue ----
#pragma unroll
  for (int s = 0; s < 2; ++s) {
#pragma unroll
    for (int t = 0; t < 2; ++t) {
      const v8f accv = (s == 0) ? (t == 0 ? acc00 : acc01) : (t == 0 ? acc10 : acc11);
      const int n = tileN + (wn * 2 + t) * 16 + l16;
      const float sc = AFFINE ? scale[n] : 1.f;
      const float bi = AFFINE ? bias[n]  : 0.f;
#pragma unroll
      for (int r = 0; r < 8; ++r) {
        const int m = tileM + (wm * 2 + s) * 16 + r + 8 * half;
        float v = accv[r] * sc + bi;
        if (HASRES) v += Rb[(long long)m * ldres + n];
        if (ACT == 1)      v = v > 0.f ? v : 0.f;
        else if (ACT == 2) v = (v > 0.f ? v : (__expf(v) - 1.f)) + 1.f;
        Cb[(long long)m * ldc + n] = v;
      }
    }
  }
}

// ===========================================================================
// Implicit-GEMM 3x3 SAME conv, pos-major [b,H*W,Cin] -> [b,H*W,F]
//   K = 9*CIN with k = tap*CIN + c  (CIN power of two -> no division)
//   ADDPE: add sine positional encoding after activation (relu)
// ===========================================================================
template<bool HASRES, bool ADDPE, int CIN>
__global__ __launch_bounds__(256) void conv3x3_wmma(
    const float* __restrict__ A, long long aBatch,
    const float* __restrict__ Wt,
    float* __restrict__ C, long long cBatch,
    const float* __restrict__ scale, const float* __restrict__ bias,
    const float* __restrict__ res, long long resBatch)
{
  constexpr int SHIFT = (CIN == 128) ? 7 : 8;
  constexpr int KTOT  = 9 * CIN;

  __shared__ alignas(32) _Float16 lA[8 * 32 * 16];
  __shared__ alignas(32) _Float16 lB[4 * 32 * 16];

  const int b     = blockIdx.z;
  const int tileM = blockIdx.x * BM;
  const int tileN = blockIdx.y * BN;

  const float* Ab = A + (long long)b * aBatch;
  float*       Cb = C + (long long)b * cBatch;
  const float* Rb = HASRES ? res + (long long)b * resBatch : nullptr;

  const int tid  = threadIdx.x;
  const int lane = tid & 31;
  const int wid  = tid >> 5;
  const int wm   = wid >> 1;
  const int wn   = wid & 1;
  const int half = lane >> 4;
  const int l16  = lane & 15;

  v8f acc00 = {}, acc01 = {}, acc10 = {}, acc11 = {};

  const int arow = tid >> 1;
  const int ako  = (tid & 1) * 16;
  const int slab = arow >> 4, row16 = arow & 15;
  const int p0   = tileM + arow;
  const int y0   = p0 >> 7;              // IMG_W = 128
  const int x0   = p0 & 127;

  const int bn  = tid >> 2;
  const int bkc = (tid & 3) * 8;
  const int bnsub = bn >> 4;
  const int blane = 16 * ((bkc >> 3) & 1) + (bn & 15);
  const int bj0   = (bkc & 16) >> 1;
  const long long wrow = (long long)(tileN + bn) * CIN;

  for (int kk = 0; kk < KTOT; kk += BK) {
    // ---- stage A: 16 contiguous channels of one shifted pixel ----
    {
      const int kg0  = kk + ako;
      const int tap  = kg0 >> SHIFT;           // constant over the 16-chunk
      const int cb   = kg0 & (CIN - 1);
      const int ys   = y0 + tap / 3 - 1;
      const int xs   = x0 + (tap % 3) - 1;
      const bool inb = (ys >= 0) & (ys < IMG_H) & (xs >= 0) & (xs < IMG_W);
      const float* src = Ab + (long long)(ys * IMG_W + xs) * CIN + cb;
      const int g = ako >> 4;
      v8h h0, h1;
#pragma unroll
      for (int i = 0; i < 8; ++i) h0[i] = (_Float16)(inb ? src[i] : 0.f);
#pragma unroll
      for (int i = 0; i < 8; ++i) h1[i] = (_Float16)(inb ? src[8 + i] : 0.f);
      *(v8h*)&lA[((slab * 32 + row16)      * 16) + 8 * g] = h0;
      *(v8h*)&lA[((slab * 32 + 16 + row16) * 16) + 8 * g] = h1;
    }
    // ---- stage B: weights [o][i][3][3], strided by 9 over channels ----
    {
      const int kg  = kk + bkc;
      const int tap = kg >> SHIFT;             // constant over the 8-chunk
      const int cb  = kg & (CIN - 1);
      const float* wsrc = Wt + (wrow + cb) * 9 + tap;
      v8h hb;
#pragma unroll
      for (int i = 0; i < 8; ++i) hb[i] = (_Float16)wsrc[i * 9];
      *(v8h*)&lB[((bnsub * 32 + blane) * 16) + bj0] = hb;
    }
    __syncthreads();

    const v16h a0 = *(const v16h*)&lA[(((wm * 2 + 0) * 32) + lane) * 16];
    const v16h a1 = *(const v16h*)&lA[(((wm * 2 + 1) * 32) + lane) * 16];
    const v16h b0 = *(const v16h*)&lB[(((wn * 2 + 0) * 32) + lane) * 16];
    const v16h b1 = *(const v16h*)&lB[(((wn * 2 + 1) * 32) + lane) * 16];
    acc00 = __builtin_amdgcn_wmma_f32_16x16x32_f16(false, a0, false, b0, (short)0, acc00, false, false);
    acc01 = __builtin_amdgcn_wmma_f32_16x16x32_f16(false, a0, false, b1, (short)0, acc01, false, false);
    acc10 = __builtin_amdgcn_wmma_f32_16x16x32_f16(false, a1, false, b0, (short)0, acc10, false, false);
    acc11 = __builtin_amdgcn_wmma_f32_16x16x32_f16(false, a1, false, b1, (short)0, acc11, false, false);
    __syncthreads();
  }

#pragma unroll
  for (int s = 0; s < 2; ++s) {
#pragma unroll
    for (int t = 0; t < 2; ++t) {
      const v8f accv = (s == 0) ? (t == 0 ? acc00 : acc01) : (t == 0 ? acc10 : acc11);
      const int n = tileN + (wn * 2 + t) * 16 + l16;
      const float sc = scale[n];
      const float bi = bias[n];
#pragma unroll
      for (int r = 0; r < 8; ++r) {
        const int m = tileM + (wm * 2 + s) * 16 + r + 8 * half;
        float v = accv[r] * sc + bi;
        if (HASRES) v += Rb[(long long)m * FDIM + n];
        v = v > 0.f ? v : 0.f;                           // relu (all conv uses)
        if (ADDPE) {
          const int yy = m >> 7, xx = m & 127;
          const int ii = n >> 2, q = n & 3;
          const float dv = __expf((float)(2 * ii) * (-9.210340371976184f / 64.f));
          const float pos = (q < 2) ? (float)(xx + 1) : (float)(yy + 1);
          const float ang = pos * dv;
          v += (q & 1) ? __cosf(ang) : __sinf(ang);
        }
        Cb[(long long)m * FDIM + n] = v;
      }
    }
  }
}

// ===========================================================================
// Elementwise / reduction helpers
// ===========================================================================
__global__ void pack_node_kernel(const float* __restrict__ node_feat,
                                 const float* __restrict__ pt_proxy,
                                 const int*   __restrict__ node2proxy,
                                 float* __restrict__ cat)
{
  const long long idx = (long long)blockIdx.x * blockDim.x + threadIdx.x;
  const long long total = (long long)BDIM * LSEQ * FDIM;
  if (idx >= total) return;
  const int f = (int)(idx % FDIM);
  const long long rest = idx / FDIM;
  const int p = (int)(rest % LSEQ);
  const int b = (int)(rest / LSEQ);
  const int pi = node2proxy[(long long)b * LSEQ + p];
  float* row = cat + ((long long)b * LSEQ + p) * (2 * FDIM);
  row[f]        = node_feat[((long long)b * FDIM + f) * LSEQ + p];
  row[FDIM + f] = pt_proxy[((long long)b * PPT + pi) * FDIM + f];
}

__global__ void pack_img_kernel(const float* __restrict__ img_feat_2,
                                const float* __restrict__ img_proxy,
                                float* __restrict__ cat)
{
  const long long idx = (long long)blockIdx.x * blockDim.x + threadIdx.x;
  const long long total = (long long)BDIM * LSEQ * FDIM;
  if (idx >= total) return;
  const int f = (int)(idx % FDIM);
  const long long rest = idx / FDIM;
  const int p = (int)(rest % LSEQ);
  const int b = (int)(rest / LSEQ);
  const int y = p >> 7, x = p & 127;
  const int pi = (y >> 2) * 32 + (x >> 2);
  float* row = cat + ((long long)b * LSEQ + p) * (2 * FDIM);
  row[f]        = img_feat_2[((long long)b * FDIM + f) * LSEQ + p];
  row[FDIM + f] = img_proxy[((long long)b * PIMG + pi) * FDIM + f];
}

__global__ void colsum_kernel(const float* __restrict__ K, float* __restrict__ Ksum)
{
  const int t = blockIdx.x * blockDim.x + threadIdx.x;
  if (t >= BDIM * FDIM) return;
  const int b = t / FDIM, f = t % FDIM;
  const float* base = K + (long long)b * LSEQ * FDIM + f;
  float s = 0.f;
  for (int l = 0; l < LSEQ; ++l) s += base[(long long)l * FDIM];
  Ksum[t] = s;
}

__global__ void kv_kernel(const float* __restrict__ K, const float* __restrict__ V,
                          float* __restrict__ KV)
{
  const int bh = blockIdx.x;
  const int b = bh / NHEADS, h = bh % NHEADS;
  const int dd = threadIdx.x >> 4, vv = threadIdx.x & 15;
  const float* kb = K + (long long)b * LSEQ * FDIM + h * HDIM + dd;
  const float* vb = V + (long long)b * LSEQ * FDIM + h * HDIM + vv;
  float acc = 0.f;
  for (int s = 0; s < LSEQ; ++s)
    acc += kb[(long long)s * FDIM] * vb[(long long)s * FDIM];
  KV[((long long)bh * HDIM + dd) * HDIM + vv] = acc * (1.f / (float)LSEQ);
}

__global__ void zmsg_kernel(const float* __restrict__ Q, const float* __restrict__ KV,
                            const float* __restrict__ Ksum, float* __restrict__ msg)
{
  __shared__ float q[FDIM];
  const int bl = blockIdx.x;
  const int b = bl / LSEQ, l = bl % LSEQ;
  const int c = threadIdx.x;
  const long long row = ((long long)b * LSEQ + l) * FDIM;
  q[c] = Q[row + c];
  __syncthreads();
  const int h = c >> 4, vv = c & 15;
  const float* kvh = KV + ((long long)b * NHEADS + h) * (HDIM * HDIM);
  const float* ksh = Ksum + (long long)b * FDIM + h * HDIM;
  float z = 0.f, m = 0.f;
#pragma unroll
  for (int d = 0; d < HDIM; ++d) {
    const float qd = q[h * HDIM + d];
    z += qd * ksh[d];
    m += qd * kvh[d * HDIM + vv];
  }
  msg[row + c] = m * (1.f / (z + 1e-6f)) * (float)LSEQ;
}

template<bool HASRES>
__global__ __launch_bounds__(256) void ln_kernel(
    const float* __restrict__ X, const float* __restrict__ g,
    const float* __restrict__ bt, const float* __restrict__ res,
    float* __restrict__ out, long long rows)
{
  const long long row = (long long)blockIdx.x * 8 + (threadIdx.x >> 5);
  const int lane = threadIdx.x & 31;
  if (row >= rows) return;
  const float* x = X + row * FDIM;
  float v[4];
  float s = 0.f;
#pragma unroll
  for (int i = 0; i < 4; ++i) { v[i] = x[lane + 32 * i]; s += v[i]; }
#pragma unroll
  for (int off = 16; off > 0; off >>= 1) s += __shfl_xor(s, off, 32);
  const float mean = s * (1.f / FDIM);
  float var = 0.f;
#pragma unroll
  for (int i = 0; i < 4; ++i) { const float d = v[i] - mean; var += d * d; }
#pragma unroll
  for (int off = 16; off > 0; off >>= 1) var += __shfl_xor(var, off, 32);
  const float inv = rsqrtf(var * (1.f / FDIM) + 1e-5f);
#pragma unroll
  for (int i = 0; i < 4; ++i) {
    const int c = lane + 32 * i;
    float y = (v[i] - mean) * inv * g[c] + bt[c];
    if (HASRES) y += res[row * FDIM + c];
    out[row * FDIM + c] = y;
  }
}

__global__ void out_kernel(const float* __restrict__ imgA,
                           const float* __restrict__ nodeA,
                           float* __restrict__ out)
{
  const long long idx = (long long)blockIdx.x * blockDim.x + threadIdx.x;
  const long long total = (long long)BDIM * FDIM * LSEQ;
  if (idx >= total) return;
  const int p = (int)(idx % LSEQ);
  const long long rest = idx / LSEQ;
  const int f = (int)(rest % FDIM);
  const int b = (int)(rest / FDIM);
  const long long src = ((long long)b * LSEQ + p) * FDIM + f;
  out[idx]         = imgA[src];
  out[total + idx] = nodeA[src];
}

// ===========================================================================
// Host orchestration
// ===========================================================================
extern "C" void kernel_launch(void* const* d_in, const int* in_sizes, int n_in,
                              void* d_out, int out_size, void* d_ws, size_t ws_size,
                              hipStream_t stream)
{
  (void)in_sizes; (void)n_in; (void)out_size; (void)ws_size;
  auto F = [&](int i) { return (const float*)d_in[i]; };

  // flattened input indices: top-level declared order, nested pytrees sorted.
  const float* img_feat_2 = F(0);
  const float* node_feat  = F(1);
  const float* img_proxy  = F(2);
  const float* pt_proxy   = F(3);
  const int*   node2proxy = (const int*)d_in[4];
  // node_fuse blk0 @5: b1 b2 bs s1 s2 ss w1 w2 ws ; blk1 @14: b1 b2 s1 s2 w1 w2
  // img_fuse  blk0 @20 ; blk1 @29
  // la @35: (n2p,nself,p2n,pself) x 2 x (bt1,bt2,g1,g2,wf1,wf2,wk,wm,wq,wv)
  auto laBase = [&](int nameIdx, int layer) { return 35 + nameIdx * 20 + layer * 10; };

  float* w = (float*)d_ws;
  size_t off = 0;
  auto alloc = [&](size_t n) { float* p = w + off; off += n; return p; };
  const size_t LF = (size_t)BDIM * LSEQ * FDIM;
  float* nodeA = alloc(LF);
  float* imgA  = alloc(LF);
  float* catb  = alloc(2 * LF);
  float* t2F   = alloc(2 * LF);
  float* tQ    = alloc(LF);
  float* tK    = alloc(LF);
  float* tV    = alloc(LF);
  float* tD    = alloc(LF);
  float* tM    = alloc(LF);
  float* bufS  = alloc(LF);
  float* KVb   = alloc((size_t)BDIM * NHEADS * HDIM * HDIM);
  float* KsumB = alloc((size_t)BDIM * FDIM);

  const long long sF  = (long long)LSEQ * FDIM;
  const long long s2F = (long long)LSEQ * 2 * FDIM;

  // generic launcher: kernel chosen at compile time per call-site
  auto G = [&](auto kfn,
               const float* A, long long aB, int lda,
               const float* A2, long long a2B, int lda2, int ksplit,
               const float* Wt, int ldb,
               float* C, long long cB, int ldc,
               int N, int K,
               const float* sc, const float* bi,
               const float* rs, long long rB, int ldr) {
    dim3 grid(LSEQ / BM, N / BN, BDIM);
    kfn<<<grid, dim3(256), 0, stream>>>(A, aB, lda, A2, a2B, lda2, ksplit,
                                        Wt, ldb, C, cB, ldc, K, sc, bi, rs, rB, ldr);
  };
  auto CV = [&](auto kfn, const float* A, long long aB, const float* Wt,
                float* C, const float* sc, const float* bi, const float* rs) {
    dim3 grid(LSEQ / BM, FDIM / BN, BDIM);
    kfn<<<grid, dim3(256), 0, stream>>>(A, aB, Wt, C, sF, sc, bi, rs, sF);
  };
  auto LN = [&](auto kfn, const float* X, const float* g, const float* bt,
                const float* rs, float* out) {
    const long long rows = (long long)BDIM * LSEQ;
    kfn<<<dim3((unsigned)(rows / 8)), dim3(256), 0, stream>>>(X, g, bt, rs, out, rows);
  };

  const long long totPF = (long long)BDIM * LSEQ * FDIM;
  const unsigned eb = (unsigned)((totPF + 255) / 256);

  // ===================== node fuse =====================
  pack_node_kernel<<<eb, 256, 0, stream>>>(node_feat, pt_proxy, node2proxy, catb);
  G(gemm_wmma<true,true,false,false,1>, catb, s2F, 2*FDIM, nullptr,0,0,0,
    F(11), 2*FDIM, tQ, sF, FDIM, FDIM, 2*FDIM, F(8), F(5), nullptr,0,0);
  G(gemm_wmma<true,true,false,false,0>, catb, s2F, 2*FDIM, nullptr,0,0,0,
    F(13), 2*FDIM, bufS, sF, FDIM, FDIM, 2*FDIM, F(10), F(7), nullptr,0,0);
  G(gemm_wmma<true,true,true,false,1>,  tQ, sF, FDIM, nullptr,0,0,0,
    F(12), FDIM, nodeA, sF, FDIM, FDIM, FDIM, F(9), F(6), bufS, sF, FDIM);
  G(gemm_wmma<true,true,false,false,1>, nodeA, sF, FDIM, nullptr,0,0,0,
    F(18), FDIM, tQ, sF, FDIM, FDIM, FDIM, F(16), F(14), nullptr,0,0);
  G(gemm_wmma<true,true,true,false,1>,  tQ, sF, FDIM, nullptr,0,0,0,
    F(19), FDIM, nodeA, sF, FDIM, FDIM, FDIM, F(17), F(15), nodeA, sF, FDIM);

  // ===================== img fuse =====================
  pack_img_kernel<<<eb, 256, 0, stream>>>(img_feat_2, img_proxy, catb);
  CV(conv3x3_wmma<false,false,256>, catb, s2F, F(26), tQ, F(23), F(20), nullptr);
  G(gemm_wmma<true,true,false,false,0>, catb, s2F, 2*FDIM, nullptr,0,0,0,
    F(28), 2*FDIM, bufS, sF, FDIM, FDIM, 2*FDIM, F(25), F(22), nullptr,0,0);
  CV(conv3x3_wmma<true,true,128>,   tQ, sF, F(27), imgA, F(24), F(21), bufS);
  CV(conv3x3_wmma<false,false,128>, imgA, sF, F(33), tQ, F(31), F(29), nullptr);
  CV(conv3x3_wmma<true,false,128>,  tQ, sF, F(34), imgA, F(32), F(30), imgA);

  // ===================== linear attention =====================
  auto la_apply = [&](float* x, const float* src, int base) {
    const float* bt1 = F(base + 0); const float* bt2 = F(base + 1);
    const float* g1  = F(base + 2); const float* g2  = F(base + 3);
    const float* wf1 = F(base + 4); const float* wf2 = F(base + 5);
    const float* wk  = F(base + 6); const float* wm  = F(base + 7);
    const float* wq  = F(base + 8); const float* wv  = F(base + 9);
    G(gemm_wmma<false,false,false,false,2>, x,   sF, FDIM, nullptr,0,0,0,
      wq, FDIM, tQ, sF, FDIM, FDIM, FDIM, nullptr,nullptr,nullptr,0,0);
    G(gemm_wmma<false,false,false,false,2>, src, sF, FDIM, nullptr,0,0,0,
      wk, FDIM, tK, sF, FDIM, FDIM, FDIM, nullptr,nullptr,nullptr,0,0);
    G(gemm_wmma<false,false,false,false,0>, src, sF, FDIM, nullptr,0,0,0,
      wv, FDIM, tV, sF, FDIM, FDIM, FDIM, nullptr,nullptr,nullptr,0,0);
    colsum_kernel<<<dim3((BDIM * FDIM + 255) / 256), 256, 0, stream>>>(tK, KsumB);
    kv_kernel<<<dim3(BDIM * NHEADS), dim3(256), 0, stream>>>(tK, tV, KVb);
    zmsg_kernel<<<dim3(BDIM * LSEQ), dim3(FDIM), 0, stream>>>(tQ, KVb, KsumB, tD);
    G(gemm_wmma<false,false,false,false,0>, tD, sF, FDIM, nullptr,0,0,0,
      wm, FDIM, tM, sF, FDIM, FDIM, FDIM, nullptr,nullptr,nullptr,0,0);
    LN(ln_kernel<false>, tM, g1, bt1, nullptr, tD);
    G(gemm_wmma<false,false,false,true,1>, x, sF, FDIM, tD, sF, FDIM, FDIM,
      wf1, 2*FDIM, t2F, s2F, 2*FDIM, 2*FDIM, 2*FDIM, nullptr,nullptr,nullptr,0,0);
    G(gemm_wmma<false,false,false,false,0>, t2F, s2F, 2*FDIM, nullptr,0,0,0,
      wf2, FDIM, tM, sF, FDIM, FDIM, 2*FDIM, nullptr,nullptr,nullptr,0,0);
    LN(ln_kernel<true>, tM, g2, bt2, x, x);
  };

  for (int i = 0; i < 2; ++i) {
    la_apply(nodeA, imgA,  laBase(2, i));   // p2n
    la_apply(imgA,  nodeA, laBase(0, i));   // n2p
    la_apply(nodeA, nodeA, laBase(1, i));   // nself
    la_apply(imgA,  imgA,  laBase(3, i));   // pself
  }

  out_kernel<<<eb, 256, 0, stream>>>(imgA, nodeA, (float*)d_out);
}
